// GraILConv_69243462746541
// MI455X (gfx1250) — matrix operand order repaired and checked by reading, not compiled, not run on hardware
//
#include <hip/hip_runtime.h>
#include <hip/hip_bf16.h>
#include <math.h>

#define DN 128   // node feature dim (K)
#define DO 128   // output dim (N of GEMM)
#define DE 32    // edge embed dim
#define RR 16    // relations
#define BB 8     // bases

typedef __bf16 bf16_t;
typedef __bf16 v16bf __attribute__((ext_vector_type(16)));
typedef __bf16 v8bf  __attribute__((ext_vector_type(8)));
typedef __bf16 v4bf  __attribute__((ext_vector_type(4)));
typedef float  v8f   __attribute__((ext_vector_type(8)));
typedef float  v4f   __attribute__((ext_vector_type(4)));

// ---------------------------------------------------------------------------
// WMMA helpers (CDNA5 wave32, v_wmma_f32_16x16x32_bf16)
// ---------------------------------------------------------------------------
__device__ __forceinline__ v8f wmma_bf16(v16bf a, v16bf b, v8f c) {
  // (neg_a, A, neg_b, B, c_mod, C, reuse_a, reuse_b)
  return __builtin_amdgcn_wmma_f32_16x16x32_bf16(false, a, false, b, (short)0, c,
                                                 false, false);
}

// A-matrix fragment (16x32 bf16, ISA 7.12.2): lane L holds row M=L%16.
// lanes 0-15: K = {0..7, 16..23}; lanes 16-31: K = {8..15, 24..31} (+kt*32)
__device__ __forceinline__ v16bf load_a_frag(const bf16_t* __restrict__ row,
                                             int kt, int lane) {
  const int kbase = kt * 32 + ((lane & 16) ? 8 : 0);
  v8bf lo = *reinterpret_cast<const v8bf*>(row + kbase);        // K=kbase..+7
  v8bf hi = *reinterpret_cast<const v8bf*>(row + kbase + 16);   // K=kbase+16..+23
  v16bf a;
#pragma unroll
  for (int j = 0; j < 8; ++j) { a[j] = lo[j]; a[j + 8] = hi[j]; }
  return a;
}

// ---------------------------------------------------------------------------
// 1) f32 -> bf16 convert of node features (4 elems / thread)
// ---------------------------------------------------------------------------
__global__ void k_cvt_bf16(const float* __restrict__ in, bf16_t* __restrict__ out,
                           int n4) {
  int tid = blockIdx.x * blockDim.x + threadIdx.x;
  if (tid < n4) {
    v4f f = *reinterpret_cast<const v4f*>(in + (size_t)tid * 4);
    v4bf b;
#pragma unroll
    for (int j = 0; j < 4; ++j) b[j] = (bf16_t)f[j];
    *reinterpret_cast<v4bf*>(out + (size_t)tid * 4) = b;
  }
}

// ---------------------------------------------------------------------------
// 2) attention alphas: only edges 0..15 are ever consumed (alphas[rels])
// ---------------------------------------------------------------------------
__global__ void k_alpha(const float* __restrict__ vfts, const int* __restrict__ adjs,
                        const float* __restrict__ er, const float* __restrict__ ert,
                        const float* __restrict__ w1, const float* __restrict__ b1,
                        const float* __restrict__ w2, const float* __restrict__ b2,
                        float* __restrict__ alpha16, int E) {
  __shared__ float red[128];
  const int t = threadIdx.x;  // 0..127
  for (int e = 0; e < RR; ++e) {
    const int s = adjs[e];
    const int d = adjs[E + e];
    float acc = b1[t];
    for (int k = 0; k < DN; ++k) acc += vfts[(size_t)s * DN + k] * w1[k * DO + t];
    for (int k = 0; k < DN; ++k) acc += vfts[(size_t)d * DN + k] * w1[(DN + k) * DO + t];
    for (int k = 0; k < DE; ++k) acc += er[e * DE + k]  * w1[(2 * DN + k) * DO + t];
    for (int k = 0; k < DE; ++k) acc += ert[e * DE + k] * w1[(2 * DN + DE + k) * DO + t];
    float h = fmaxf(acc, 0.f);          // relu
    red[t] = h * w2[t];
    __syncthreads();
    for (int s2 = 64; s2 > 0; s2 >>= 1) {
      if (t < s2) red[t] += red[t + s2];
      __syncthreads();
    }
    if (t == 0) alpha16[e] = 1.f / (1.f + expf(-(red[0] + b2[0])));
    __syncthreads();
  }
}

// ---------------------------------------------------------------------------
// 3) build alpha-scaled relation weights as pre-swizzled bf16 B-fragments
//    layout: frag[((r*8 + nt)*4 + kt)*512 + lane*16 + j]
//    B (KxN) per lane: n = nt*16 + lane%16 ; k = kt*32 + (lane<16?0:8) + (j&7) + (j<8?0:16)
// ---------------------------------------------------------------------------
__global__ void k_build_wrel(const float* __restrict__ weight,
                             const float* __restrict__ comp,
                             const float* __restrict__ alpha16,
                             bf16_t* __restrict__ frag) {
  int tid = blockIdx.x * blockDim.x + threadIdx.x;
  if (tid >= RR * 8 * 4 * 32 * 16) return;
  const int j    = tid & 15;
  const int lane = (tid >> 4) & 31;
  const int kt   = (tid >> 9) & 3;
  const int nt   = (tid >> 11) & 7;
  const int r    = tid >> 14;
  const int k = kt * 32 + ((lane & 16) ? 8 : 0) + (j & 7) + ((j & 8) ? 16 : 0);
  const int n = nt * 16 + (lane & 15);
  float acc = 0.f;
#pragma unroll
  for (int b = 0; b < BB; ++b)
    acc += comp[r * BB + b] * weight[((size_t)b * DN + k) * DO + n];
  acc *= alpha16[r];  // fold per-relation attention scalar into the weights
  frag[tid] = (bf16_t)acc;
}

__global__ void k_build_upd(const float* __restrict__ update,
                            bf16_t* __restrict__ frag) {
  int tid = blockIdx.x * blockDim.x + threadIdx.x;
  if (tid >= 8 * 4 * 32 * 16) return;
  const int j    = tid & 15;
  const int lane = (tid >> 4) & 31;
  const int kt   = (tid >> 9) & 3;
  const int nt   = tid >> 11;
  const int k = kt * 32 + ((lane & 16) ? 8 : 0) + (j & 7) + ((j & 8) ? 16 : 0);
  const int n = nt * 16 + (lane & 15);
  frag[tid] = (bf16_t)update[(size_t)k * DO + n];
}

// ---------------------------------------------------------------------------
// 4) counting sort by relation + in-degree histogram
// ---------------------------------------------------------------------------
__global__ void k_hist(const int* __restrict__ rels, const int* __restrict__ adjs,
                       int* __restrict__ bins, int* __restrict__ indeg, int E) {
  int i = blockIdx.x * blockDim.x + threadIdx.x;
  if (i < E) {
    atomicAdd(&bins[rels[i]], 1);
    atomicAdd(&indeg[adjs[E + i]], 1);   // dst = adjs[1][i]
  }
}

__global__ void k_prefix(const int* __restrict__ bins, int* __restrict__ cursor) {
  if (threadIdx.x == 0 && blockIdx.x == 0) {
    int run = 0;
    for (int r = 0; r < RR; ++r) { cursor[r] = run; run += bins[r]; }
  }
}

__global__ void k_scatter(const int* __restrict__ rels, int* __restrict__ cursor,
                          int* __restrict__ perm, int E) {
  int i = blockIdx.x * blockDim.x + threadIdx.x;
  if (i < E) {
    int pos = atomicAdd(&cursor[rels[i]], 1);
    perm[pos] = i;
  }
}

// ---------------------------------------------------------------------------
// 5) dsts = vfts @ update   [N,128]x[128,128], WMMA bf16 -> f32
//    block = 256 thr = 8 waves; block owns 16 rows; wave w owns N-tile w.
//    Two independent accumulator chains hide the WMMA RAW hazard.
// ---------------------------------------------------------------------------
__global__ __launch_bounds__(256) void k_dsts(const bf16_t* __restrict__ vfb,
                                              const bf16_t* __restrict__ updf,
                                              float* __restrict__ dsts, int N) {
  const int t = threadIdx.x, l = t & 31, nt = t >> 5;
  const int m0 = blockIdx.x * 16;
  int mrow = m0 + (l & 15);
  if (mrow >= N) mrow = N - 1;   // tail clamp (loads only)
  const bf16_t* row = vfb + (size_t)mrow * DN;
  const bf16_t* bp  = updf + nt * 4 * 512 + l * 16;

  v8f c0 = {}, c1 = {};
#pragma unroll
  for (int kt = 0; kt < 4; kt += 2) {
    v16bf a0 = load_a_frag(row, kt, l);
    v16bf a1 = load_a_frag(row, kt + 1, l);
    v16bf b0 = *reinterpret_cast<const v16bf*>(bp + kt * 512);
    v16bf b1 = *reinterpret_cast<const v16bf*>(bp + (kt + 1) * 512);
    c0 = wmma_bf16(a0, b0, c0);
    c1 = wmma_bf16(a1, b1, c1);
  }
  v8f c = c0 + c1;

  // C layout: VGPR v -> M = m0 + v + (lane>=16 ? 8 : 0), N = nt*16 + lane%16
  const int n  = nt * 16 + (l & 15);
  const int mb = m0 + ((l & 16) ? 8 : 0);
  if (m0 + 16 <= N) {            // uniform fast path: unconditional stores
#pragma unroll
    for (int v = 0; v < 8; ++v) dsts[(size_t)(mb + v) * DO + n] = c[v];
  } else {
#pragma unroll
    for (int v = 0; v < 8; ++v) {
      const int m = mb + v;
      if (m < N) dsts[(size_t)m * DO + n] = c[v];
    }
  }
}

// ---------------------------------------------------------------------------
// 6) per-edge GEMM (rel-sorted 16-edge tiles) + atomic scatter into out
// ---------------------------------------------------------------------------
__global__ __launch_bounds__(256) void k_edge(const bf16_t* __restrict__ vfb,
                                              const bf16_t* __restrict__ wrelf,
                                              const int* __restrict__ perm,
                                              const int* __restrict__ rels,
                                              const int* __restrict__ adjs,
                                              float* __restrict__ out, int E) {
  __shared__ int s_src[16], s_dst[16], s_rel[16];
  const int t = threadIdx.x;
  const int base = blockIdx.x * 16;
  if (t < 16) {
    const int gi = base + t;
    if (gi < E) {
      const int e = perm[gi];
      s_src[t] = adjs[e];
      s_dst[t] = adjs[E + e];
      s_rel[t] = rels[e];
    } else {
      s_src[t] = 0; s_dst[t] = 0; s_rel[t] = -1;  // padded lanes never match any r
    }
  }
  __syncthreads();

  const int l = t & 31, nt = t >> 5;
  const int i = l & 15;                       // matrix row (edge slot) this lane feeds
  const bf16_t* row = vfb + (size_t)s_src[i] * DN;
  v16bf a[4];
#pragma unroll
  for (int kt = 0; kt < 4; ++kt) a[kt] = load_a_frag(row, kt, l);

  const int  myrel = s_rel[i];
  const bool full  = (base + 16 <= E);
  const int  rlo   = s_rel[0];                       // tiles are rel-sorted
  const int  rhi   = full ? s_rel[15] : (RR - 1);

  v8f c0 = {}, c1 = {};
  if (full && rlo == rhi) {                          // fast path: single relation
    const bf16_t* bp = wrelf + (size_t)((rlo * 8 + nt) * 4) * 512 + l * 16;
#pragma unroll
    for (int kt = 0; kt < 4; kt += 2) {
      v16bf b0 = *reinterpret_cast<const v16bf*>(bp + kt * 512);
      v16bf b1 = *reinterpret_cast<const v16bf*>(bp + (kt + 1) * 512);
      c0 = wmma_bf16(a[kt], b0, c0);
      c1 = wmma_bf16(a[kt + 1], b1, c1);
    }
  } else {                                           // mixed tile: mask rows per rel
    v16bf zero = {};
    for (int r = rlo; r <= rhi; ++r) {
      const bool on = (myrel == r);
      const bf16_t* bp = wrelf + (size_t)((r * 8 + nt) * 4) * 512 + l * 16;
#pragma unroll
      for (int kt = 0; kt < 4; kt += 2) {
        v16bf am0 = on ? a[kt]     : zero;           // lane-uniform row mask
        v16bf am1 = on ? a[kt + 1] : zero;
        v16bf b0  = *reinterpret_cast<const v16bf*>(bp + kt * 512);
        v16bf b1  = *reinterpret_cast<const v16bf*>(bp + (kt + 1) * 512);
        c0 = wmma_bf16(am0, b0, c0);
        c1 = wmma_bf16(am1, b1, c1);
      }
    }
  }
  v8f c = c0 + c1;

  // scatter: C row v -> edge slot (v + (lane>=16?8:0)); alpha already folded in B
  const int n  = nt * 16 + i;
  const int ib = (l & 16) ? 8 : 0;
#pragma unroll
  for (int v = 0; v < 8; ++v) {
    unsafeAtomicAdd(&out[(size_t)s_dst[ib + v] * DO + n], c[v]);
  }
}

// ---------------------------------------------------------------------------
// 7) out = relu(acc + indeg * dsts)
// ---------------------------------------------------------------------------
__global__ void k_final(float* __restrict__ out, const float* __restrict__ dsts,
                        const int* __restrict__ indeg, int n) {
  int tid = blockIdx.x * blockDim.x + threadIdx.x;
  if (tid < n) {
    const int node = tid >> 7;  // /128
    float v = out[tid] + (float)indeg[node] * dsts[tid];
    out[tid] = fmaxf(v, 0.f);
  }
}

// ---------------------------------------------------------------------------
extern "C" void kernel_launch(void* const* d_in, const int* in_sizes, int n_in,
                              void* d_out, int out_size, void* d_ws, size_t ws_size,
                              hipStream_t stream) {
  const float* vfts   = (const float*)d_in[0];
  const int*   adjs   = (const int*)d_in[1];
  const int*   rels   = (const int*)d_in[2];
  const float* er     = (const float*)d_in[3];
  const float* ert    = (const float*)d_in[4];
  const float* weight = (const float*)d_in[5];
  const float* comp   = (const float*)d_in[6];
  const float* w1     = (const float*)d_in[7];
  const float* b1     = (const float*)d_in[8];
  const float* w2     = (const float*)d_in[9];
  const float* b2     = (const float*)d_in[10];
  const float* update = (const float*)d_in[11];
  float* out = (float*)d_out;

  const int N = in_sizes[0] / DN;
  const int E = in_sizes[2];

  // workspace carve-up (256B aligned)
  char* ws = (char*)d_ws;
  size_t off = 0;
  auto take = [&](size_t bytes) -> char* {
    char* p = ws + off;
    off += (bytes + 255) & ~(size_t)255;
    return p;
  };
  bf16_t* vfb     = (bf16_t*)take((size_t)N * DN * sizeof(bf16_t));
  bf16_t* wrelf   = (bf16_t*)take((size_t)RR * DN * DO * sizeof(bf16_t));
  bf16_t* updf    = (bf16_t*)take((size_t)DN * DO * sizeof(bf16_t));
  float*  dsts    = (float*)take((size_t)N * DO * sizeof(float));
  float*  alpha16 = (float*)take(RR * sizeof(float));
  int*    indeg   = (int*)take((size_t)N * sizeof(int));
  int*    perm    = (int*)take((size_t)E * sizeof(int));
  int*    bins    = (int*)take(2 * RR * sizeof(int));  // bins[16] + cursor[16]
  int*    cursor  = bins + RR;
  (void)ws_size;

  // zero accumulators (out doubles as the atomic scatter accumulator)
  hipMemsetAsync(out, 0, (size_t)N * DO * sizeof(float), stream);
  hipMemsetAsync(indeg, 0, (size_t)N * sizeof(int), stream);
  hipMemsetAsync(bins, 0, 2 * RR * sizeof(int), stream);

  const int nfeat4 = (N * DN) / 4;   // N*DN is a multiple of 4 (DN=128)
  k_cvt_bf16<<<(nfeat4 + 255) / 256, 256, 0, stream>>>(vfts, vfb, nfeat4);

  k_alpha<<<1, 128, 0, stream>>>(vfts, adjs, er, ert, w1, b1, w2, b2, alpha16, E);

  k_build_wrel<<<(RR * DN * DO + 255) / 256, 256, 0, stream>>>(weight, comp,
                                                               alpha16, wrelf);
  k_build_upd<<<(DN * DO + 255) / 256, 256, 0, stream>>>(update, updf);

  k_hist<<<(E + 255) / 256, 256, 0, stream>>>(rels, adjs, bins, indeg, E);
  k_prefix<<<1, 32, 0, stream>>>(bins, cursor);
  k_scatter<<<(E + 255) / 256, 256, 0, stream>>>(rels, cursor, perm, E);

  k_dsts<<<(N + 15) / 16, 256, 0, stream>>>(vfb, updf, dsts, N);

  k_edge<<<(E + 15) / 16, 256, 0, stream>>>(vfb, wrelf, perm, rels, adjs, out, E);

  const int ntot = N * DO;
  k_final<<<(ntot + 255) / 256, 256, 0, stream>>>(out, dsts, indeg, ntot);
}
